// CondConv_71305047048808
// MI455X (gfx1250) — compile-verified, matrix operand order checked by reading, not run on hardware
//
#include <hip/hip_runtime.h>

// ---------------- problem constants ----------------
#define BS    32
#define CIN   256
#define HH    56
#define WW    56
#define COUT  256
#define KEXP  4
#define HP    58            // padded spatial
#define PIX   (HH*WW)       // 3136 = 98*32
#define MT2   98            // M blocks of 32
#define KTOT  (CIN*9)       // 2304
#define KS    72            // k-steps of 32 (k = r*256 + ci)
#define NG    4             // N groups of 64 channels

typedef __attribute__((ext_vector_type(16))) __bf16       v16bf;
typedef __attribute__((ext_vector_type(8)))  float        v8f;
typedef __attribute__((ext_vector_type(4)))  unsigned int u32x4;
typedef __attribute__((ext_vector_type(4)))  float        f32x4;

union ABfrag { v16bf v; u32x4 u[2]; };

__device__ __forceinline__ unsigned short f2bf(float f) {
    unsigned u = __float_as_uint(f);
    u += 0x7FFFu + ((u >> 16) & 1u);     // round-to-nearest-even
    return (unsigned short)(u >> 16);
}

// -------- kernel 1: global average pool: pooled[b*CIN+ci] --------
__global__ void pool_kernel(const float* __restrict__ x, float* __restrict__ pooled) {
    const int bc = blockIdx.x;                 // 0..8191
    const float* p = x + (size_t)bc * PIX;
    float s = 0.f;
    for (int i = threadIdx.x; i < PIX; i += 256) s += p[i];
    __shared__ float red[256];
    red[threadIdx.x] = s;
    __syncthreads();
    for (int o = 128; o > 0; o >>= 1) {
        if (threadIdx.x < o) red[threadIdx.x] += red[threadIdx.x + o];
        __syncthreads();
    }
    if (threadIdx.x == 0) pooled[bc] = red[0] * (1.0f / (float)PIX);
}

// -------- kernel 2: attention + BN scale + fused bias/BN shift --------
__global__ void att_kernel(const float* __restrict__ pooled, const float* __restrict__ att_w,
                           const float* __restrict__ bias,   const float* __restrict__ gamma,
                           const float* __restrict__ beta,   const float* __restrict__ mean,
                           const float* __restrict__ var,
                           float* __restrict__ att, float* __restrict__ inv, float* __restrict__ addt) {
    __shared__ float att_s[BS * KEXP];
    const int t = threadIdx.x;
    if (t < BS * KEXP) {
        const int b = t >> 2, e = t & 3;
        const float* pb = pooled + b * CIN;
        const float* we = att_w + e * CIN;
        float s = 0.f;
        for (int i = 0; i < CIN; ++i) s += pb[i] * we[i];
        const float a = 1.0f / (1.0f + __expf(-s));
        att_s[t] = a;
        att[t]   = a;
    }
    inv[t] = gamma[t] * rsqrtf(var[t] + 1e-5f);
    __syncthreads();
    for (int it = 0; it < BS; ++it) {
        const int idx = it * 256 + t;          // (b,co)
        const int b = idx >> 8, co = idx & 255;
        float ab = 0.f;
        for (int e = 0; e < KEXP; ++e) ab += att_s[b * 4 + e] * bias[e * COUT + co];
        const float iv = gamma[co] * rsqrtf(var[co] + 1e-5f);
        addt[idx] = (ab - mean[co]) * iv + beta[co];
    }
}

// -------- kernel 3: x (NCHW f32) -> x_pad (NHWC bf16, zero halo) --------
__global__ void pad_kernel(const float* __restrict__ x, unsigned short* __restrict__ xpad) {
    const int idx = blockIdx.x * 256 + threadIdx.x;    // ((b*58+y)*58+xc)*256+ci
    const int ci = idx & 255;
    int r = idx >> 8;
    const int xc = r % HP; r /= HP;
    const int y  = r % HP;
    const int b  = r / HP;
    float v = 0.f;
    if (y >= 1 && y <= HH && xc >= 1 && xc <= WW)
        v = x[(((size_t)(b * CIN + ci)) * HH + (y - 1)) * WW + (xc - 1)];
    xpad[idx] = f2bf(v);
}

// -------- kernel 4: expert-weight aggregation into WMMA B-fragment layout --------
// aggw layout: [b][ct(16)][ks(72)][lane(32)][16 bf16]  (1 KB per fragment)
__global__ void aggw_kernel(const float* __restrict__ weight, const float* __restrict__ att,
                            unsigned short* __restrict__ aggw) {
    const int tau = blockIdx.x * 256 + threadIdx.x;    // 16*72*32*16 = 589824
    const int j  = tau & 15;
    const int l  = (tau >> 4) & 31;
    const int ks = (tau >> 9) % KS;
    const int ct = (tau >> 9) / KS;
    const int n = l & 15, khalf = l >> 4;
    const int co = ct * 16 + n;
    const int koff = (j < 8) ? (khalf * 8 + j) : (16 + khalf * 8 + (j - 8));
    const int k  = ks * 32 + koff;                     // k = r*256 + ci
    const int r  = k >> 8;
    const int ci = k & 255;
    float w[KEXP];
    #pragma unroll
    for (int e = 0; e < KEXP; ++e)
        w[e] = weight[((size_t)((e * COUT + co) * CIN + ci)) * 9 + r];
    const f32x4* av = (const f32x4*)att;
    for (int b = 0; b < BS; ++b) {
        const f32x4 a = av[b];
        const float s = a[0] * w[0] + a[1] * w[1] + a[2] * w[2] + a[3] * w[3];
        aggw[(((size_t)(b * 16 + ct) * KS + ks) << 9) + l * 16 + j] = f2bf(s);
    }
}

// -------- kernel 5: implicit-GEMM conv via v_wmma_f32_16x16x32_bf16 --------
// wave tile: 32 (spatial) x 64 (out-channels); M=3136, N=256, K=2304
// block's 8 waves share (b, ng) => B fragments hit WGP$; A distinct per wave.
__global__ __launch_bounds__(256) void conv_kernel(const unsigned short* __restrict__ xpad,
                                                   const unsigned short* __restrict__ aggw,
                                                   const float* __restrict__ inv,
                                                   const float* __restrict__ addt,
                                                   float* __restrict__ out) {
    __shared__ float ldsT[8][64 * 20];                 // padded stride 20 -> conflict-free
    const int tid  = threadIdx.x;
    const int w    = tid >> 5;
    const int lane = tid & 31;
    const int wg   = blockIdx.x * 8 + w;               // 12544 waves total
    const int b    = wg / (MT2 * NG);
    const int rem  = wg % (MT2 * NG);
    const int ng   = rem / MT2;                        // slow within block -> B shared
    const int mt2  = rem % MT2;                        // fast within block

    const int n = lane & 15, khalf = lane >> 4;
    const int p0 = mt2 * 32 + n;                       // A rows p0 and p0+16
    const int p1 = p0 + 16;
    const int py0 = p0 / WW, px0 = p0 - py0 * WW;
    const int py1 = p1 / WW, px1 = p1 - py1 * WW;
    const int rowoff0 = (py0 * HP + px0) * CIN;
    const int rowoff1 = (py1 * HP + px1) * CIN;

    const unsigned short* xb = xpad + (size_t)b * HP * HP * CIN;

    // B fragments: base for this wave's 4 co-tiles (ct = ng*4 .. ng*4+3)
    const unsigned short* fb = aggw + (((size_t)(b * 16 + ng * 4) * KS) << 9) + lane * 16;

    v8f acc[8] = {};                                   // acc[mh*4 + s]
    for (int ks = 0; ks < KS; ++ks) {
        const int r  = ks >> 3;
        const int dy = r / 3, dx = r - 3 * (r / 3);
        const int ci0 = (ks & 7) * 32;
        const int doff = (dy * HP + dx) * CIN + ci0 + khalf * 8;

        ABfrag a0, a1;
        const unsigned short* ap0 = xb + rowoff0 + doff;
        a0.u[0] = *(const u32x4*)(ap0);
        a0.u[1] = *(const u32x4*)(ap0 + 16);
        const unsigned short* ap1 = xb + rowoff1 + doff;
        a1.u[0] = *(const u32x4*)(ap1);
        a1.u[1] = *(const u32x4*)(ap1 + 16);

        ABfrag bf[4];
        const unsigned short* bp = fb + ((size_t)ks << 9);
        #pragma unroll
        for (int s = 0; s < 4; ++s) {
            bf[s].u[0] = *(const u32x4*)(bp + (size_t)s * KS * 512);
            bf[s].u[1] = *(const u32x4*)(bp + (size_t)s * KS * 512 + 8);
        }
        #pragma unroll
        for (int s = 0; s < 4; ++s)
            acc[s] = __builtin_amdgcn_wmma_f32_16x16x32_bf16(
                false, a0.v, false, bf[s].v, (short)0, acc[s], false, false);
        #pragma unroll
        for (int s = 0; s < 4; ++s)
            acc[4 + s] = __builtin_amdgcn_wmma_f32_16x16x32_bf16(
                false, a1.v, false, bf[s].v, (short)0, acc[4 + s], false, false);
    }

    // transpose D tiles through LDS (two passes over one 64x20 region per wave)
    float* my = &ldsT[w][0];
    const int nbase = ng * 64;
    #pragma unroll
    for (int mh = 0; mh < 2; ++mh) {
        #pragma unroll
        for (int s = 0; s < 4; ++s)
            #pragma unroll
            for (int r = 0; r < 8; ++r)
                my[(s * 16 + n) * 20 + (r + khalf * 8)] = acc[mh * 4 + s][r];
        __syncthreads();

        const int mbase = mt2 * 32 + mh * 16;
        #pragma unroll
        for (int half = 0; half < 2; ++half) {
            const int col = lane + half * 32;          // 0..63 co within tile
            const int co  = nbase + col;
            const float sc = inv[co];
            const float ad = addt[b * COUT + co];
            float* ob = out + (size_t)(b * COUT + co) * PIX + mbase;
            #pragma unroll
            for (int jj = 0; jj < 4; ++jj) {
                f32x4 v = *(const f32x4*)&my[col * 20 + jj * 4];
                #pragma unroll
                for (int c = 0; c < 4; ++c) v[c] = fmaxf(fmaf(v[c], sc, ad), 0.0f);
                *(f32x4*)(ob + jj * 4) = v;
            }
        }
        __syncthreads();
    }
}

// ---------------- host launcher ----------------
extern "C" void kernel_launch(void* const* d_in, const int* in_sizes, int n_in,
                              void* d_out, int out_size, void* d_ws, size_t ws_size,
                              hipStream_t stream) {
    const float* x      = (const float*)d_in[0];
    const float* att_w  = (const float*)d_in[1];
    const float* weight = (const float*)d_in[2];
    const float* bias   = (const float*)d_in[3];
    const float* gamma  = (const float*)d_in[4];
    const float* beta   = (const float*)d_in[5];
    const float* rmean  = (const float*)d_in[6];
    const float* rvar   = (const float*)d_in[7];
    float* out = (float*)d_out;

    char* ws = (char*)d_ws;
    const size_t XPAD_BYTES = (size_t)BS * HP * HP * CIN * 2;       // 55,115,776
    unsigned short* xpad   = (unsigned short*)(ws);
    float*          att    = (float*)(ws + XPAD_BYTES);             // 512 B
    float*          pooled = (float*)(ws + XPAD_BYTES + 512);       // 32 KB
    float*          inv    = (float*)(ws + XPAD_BYTES + 512 + 32768);
    float*          addt   = (float*)(ws + XPAD_BYTES + 512 + 32768 + 1024);
    unsigned short* aggw   = (unsigned short*)(ws + XPAD_BYTES + 512 + 32768 + 1024 + 32768);

    pool_kernel<<<BS * CIN, 256, 0, stream>>>(x, pooled);
    att_kernel<<<1, 256, 0, stream>>>(pooled, att_w, bias, gamma, beta, rmean, rvar,
                                      att, inv, addt);
    pad_kernel<<<(BS * HP * HP * CIN) / 256, 256, 0, stream>>>(x, xpad);
    aggw_kernel<<<(16 * KS * 32 * 16) / 256, 256, 0, stream>>>(weight, att, aggw);
    conv_kernel<<<(BS * MT2 * NG) / 8, 256, 0, stream>>>(xpad, aggw, inv, addt, out);
}
